// BasicTransformerBlock_25718264168868
// MI455X (gfx1250) — matrix operand order repaired and checked by reading, not compiled, not run on hardware
//
#include <hip/hip_runtime.h>
#include <hip/hip_bf16.h>

// ---------------------------------------------------------------------------
// BasicTransformerBlock for MI455X (gfx1250). All GEMM-shaped work runs on
// v_wmma_f32_16x16x32_f16 (wave32). A-tile staging uses the gfx1250 async
// global->LDS instruction via inline asm (ASYNCcnt-tracked); K+2 tiles are
// prefetched via global_prefetch_b8.
// ---------------------------------------------------------------------------

typedef _Float16 half_t;
typedef __attribute__((ext_vector_type(16))) _Float16 v16h;
typedef __attribute__((ext_vector_type(8)))  float    v8f;

#if defined(__gfx1250__)
#define ASYNC_LDS 1
#else
#define ASYNC_LDS 0
#endif

// async 16-byte global->LDS copy (per-lane addresses), tracked by ASYNCcnt
__device__ __forceinline__ void async_copy_b128(const void* gsrc, void* lds_dst) {
#if ASYNC_LDS
    unsigned int lds_addr = (unsigned int)(uintptr_t)lds_dst;      // addr[31:0] = LDS offset
    unsigned long long gaddr = (unsigned long long)(uintptr_t)gsrc;
    asm volatile("global_load_async_to_lds_b128 %0, %1, off"
                 :: "v"(lds_addr), "v"(gaddr) : "memory");
#else
    *(uint4*)lds_dst = *(const uint4*)gsrc;
#endif
}

__device__ __forceinline__ void wait_async_lds() {
#if ASYNC_LDS
    asm volatile("s_wait_asynccnt 0x0" ::: "memory");
#endif
}

#define BB   2
#define SS   4096
#define DIMC 512
#define HH   8
#define DD   64
#define TT   77
#define TPAD 128

// ---------------------------------------------------------------------------
// f16 WMMA GEMM: out = A(MxK) @ W(KxN) [+bias][+resid]. 128x128 block tile,
// 8 waves (2Mx4N), 64x32 per wave. Double-buffered LDS, async A staging.
// ---------------------------------------------------------------------------
#define TBM 128
#define TBN 128
#define TBK 32
#define APAD 40
#define BPAD 40

template <bool HAS_BIAS, bool HAS_RES, bool OUT_F32>
__global__ __launch_bounds__(256)
void gemm_f16_wmma_kernel(const half_t* __restrict__ A,
                          const half_t* __restrict__ W,
                          int M, int N, int K,
                          const float* __restrict__ bias,
                          const float* __restrict__ resid,
                          float* __restrict__ outF,
                          half_t* __restrict__ outH)
{
    __shared__ half_t As[2][TBM * APAD];
    __shared__ half_t Bs[2][TBN * BPAD];

    const int t    = threadIdx.x;
    const int lane = t & 31;
    const int wave = t >> 5;
    const int wm   = wave & 1;
    const int wn   = wave >> 1;
    const int hi   = lane >> 4;
    const int m16  = lane & 15;
    const int m0   = blockIdx.y * TBM;
    const int n0   = blockIdx.x * TBN;

    // stage A tile (128x32, contiguous rows): async global->LDS
    auto stageA = [&](int k0, int buf) {
        #pragma unroll
        for (int i = 0; i < 2; ++i) {
            int e   = t + i * 256;          // 512 x 16B chunks
            int row = e >> 2;
            int seg = e & 3;
            const half_t* src = A + (size_t)(m0 + row) * K + k0 + seg * 8;
            half_t*       dst = &As[buf][row * APAD + seg * 8];
            async_copy_b128(src, dst);
        }
    };
    // stage B tile (32x128) transposed into Bs[n][k] (needs VGPR transpose)
    auto stageB = [&](int k0, int buf) {
        #pragma unroll
        for (int e = t; e < (TBK * TBN / 2); e += 256) {
            int kr = e >> 6;
            int np = e & 63;
            uint32_t v = *(const uint32_t*)(W + (size_t)(k0 + kr) * N + n0 + np * 2);
            Bs[buf][(np * 2    ) * BPAD + kr] = ((const half_t*)&v)[0];
            Bs[buf][(np * 2 + 1) * BPAD + kr] = ((const half_t*)&v)[1];
        }
    };

    v8f zero = {};
    v8f acc[4][2];
    #pragma unroll
    for (int i = 0; i < 4; ++i)
        #pragma unroll
        for (int j = 0; j < 2; ++j) acc[i][j] = zero;

    stageA(0, 0);
    stageB(0, 0);
    wait_async_lds();
    __syncthreads();

    int buf = 0;
    for (int k0 = 0; k0 < K; k0 += TBK) {
        const int kn = k0 + TBK;
        if (kn < K) { stageA(kn, buf ^ 1); stageB(kn, buf ^ 1); }
        // prefetch the K+2 tile into cache (global_prefetch_b8)
        if (k0 + 2 * TBK < K && t < 128) {
            __builtin_prefetch(A + (size_t)(m0 + t) * K + k0 + 2 * TBK, 0, 0);
            __builtin_prefetch(W + (size_t)(k0 + 2 * TBK + (t >> 2)) * N + n0 + (t & 3) * 32, 0, 0);
        }

        v16h afr[4];
        #pragma unroll
        for (int i = 0; i < 4; ++i) {
            const half_t* ap = &As[buf][(wm * 64 + i * 16 + m16) * APAD];
            v16h a;
            #pragma unroll
            for (int j = 0; j < 8; ++j) a[j]     = ap[hi * 8 + j];
            #pragma unroll
            for (int j = 0; j < 8; ++j) a[8 + j] = ap[16 + hi * 8 + j];
            afr[i] = a;
        }
        v16h bfr[2];
        #pragma unroll
        for (int jn = 0; jn < 2; ++jn) {
            const half_t* bp = &Bs[buf][(wn * 32 + jn * 16 + m16) * BPAD];
            v16h b;
            #pragma unroll
            for (int j = 0; j < 8; ++j) b[j]     = bp[hi * 8 + j];
            #pragma unroll
            for (int j = 0; j < 8; ++j) b[8 + j] = bp[16 + hi * 8 + j];
            bfr[jn] = b;
        }
        #pragma unroll
        for (int i = 0; i < 4; ++i)
            #pragma unroll
            for (int jn = 0; jn < 2; ++jn)
                acc[i][jn] = __builtin_amdgcn_wmma_f32_16x16x32_f16(
                    false, afr[i], false, bfr[jn], (short)0, acc[i][jn], false, false);

        wait_async_lds();
        __syncthreads();
        buf ^= 1;
    }

    #pragma unroll
    for (int i = 0; i < 4; ++i)
        #pragma unroll
        for (int jn = 0; jn < 2; ++jn) {
            int col = n0 + wn * 32 + jn * 16 + m16;
            float bv = HAS_BIAS ? bias[col] : 0.f;
            #pragma unroll
            for (int r = 0; r < 8; ++r) {
                int row = m0 + wm * 64 + i * 16 + hi * 8 + r;
                float v = acc[i][jn][r] + bv;
                if (HAS_RES) v += resid[(size_t)row * N + col];
                if (OUT_F32) outF[(size_t)row * N + col] = v;
                else         outH[(size_t)row * N + col] = (half_t)v;
            }
        }
}

// ---------------------------------------------------------------------------
// Flash attention: one wave per (b, h, 16-query tile), 32 keys per iteration.
// 4 score WMMAs + 4 full PV WMMAs per iteration; P transposed through LDS.
// ---------------------------------------------------------------------------
__global__ __launch_bounds__(32)
void flash_attn_wmma_kernel(const half_t* __restrict__ Q,
                            const half_t* __restrict__ Kt,
                            const half_t* __restrict__ Vt,
                            half_t* __restrict__ O,
                            int S, int Tk, int kv_len, float scale)
{
    __shared__ half_t P[16 * 32];

    const int lane = threadIdx.x & 31;
    const int hi   = lane >> 4;
    const int m16  = lane & 15;
    const int q0   = blockIdx.x * 16;
    const int h    = blockIdx.y;
    const int b    = blockIdx.z;
    const int HD   = HH * DD;

    const half_t* qb = Q  + ((size_t)b * S)  * HD + h * DD;
    const half_t* kb = Kt + ((size_t)b * Tk) * HD + h * DD;
    const half_t* vb = Vt + ((size_t)b * Tk) * HD + h * DD;

    // Q as A-operand, two K=32 chunks over d
    v16h aq[2];
    #pragma unroll
    for (int c = 0; c < 2; ++c) {
        v16h a;
        const half_t* qr = qb + (size_t)(q0 + m16) * HD + c * 32;
        #pragma unroll
        for (int j = 0; j < 8; ++j) a[j]     = qr[hi * 8 + j];
        #pragma unroll
        for (int j = 0; j < 8; ++j) a[8 + j] = qr[16 + hi * 8 + j];
        aq[c] = a;
    }

    v8f zero = {};
    v8f o[4];
    #pragma unroll
    for (int d = 0; d < 4; ++d) o[d] = zero;
    float mrow[8], lrow[8];
    #pragma unroll
    for (int r = 0; r < 8; ++r) { mrow[r] = -1e30f; lrow[r] = 0.f; }

    for (int kk = 0; kk < Tk; kk += 32) {
        if (kk + 32 < Tk) {   // prefetch next K/V rows
            __builtin_prefetch(kb + (size_t)(kk + 32 + m16) * HD, 0, 0);
            __builtin_prefetch(kb + (size_t)(kk + 48 + m16) * HD, 0, 0);
            __builtin_prefetch(vb + (size_t)(kk + 32 + m16) * HD, 0, 0);
            __builtin_prefetch(vb + (size_t)(kk + 48 + m16) * HD, 0, 0);
        }
        // scores for two 16-key tiles
        v8f s0 = zero, s1 = zero;
        #pragma unroll
        for (int c = 0; c < 2; ++c) {
            v16h bk0, bk1;
            const half_t* kr0 = kb + (size_t)(kk      + m16) * HD + c * 32;
            const half_t* kr1 = kb + (size_t)(kk + 16 + m16) * HD + c * 32;
            #pragma unroll
            for (int j = 0; j < 8; ++j) { bk0[j] = kr0[hi * 8 + j]; bk1[j] = kr1[hi * 8 + j]; }
            #pragma unroll
            for (int j = 0; j < 8; ++j) { bk0[8 + j] = kr0[16 + hi * 8 + j]; bk1[8 + j] = kr1[16 + hi * 8 + j]; }
            s0 = __builtin_amdgcn_wmma_f32_16x16x32_f16(
                false, aq[c], false, bk0, (short)0, s0, false, false);
            s1 = __builtin_amdgcn_wmma_f32_16x16x32_f16(
                false, aq[c], false, bk1, (short)0, s1, false, false);
        }

        const bool mk0 = (kk + m16)      >= kv_len;
        const bool mk1 = (kk + 16 + m16) >= kv_len;
        float p0[8], p1[8], tmax[8];
        #pragma unroll
        for (int r = 0; r < 8; ++r) {
            p0[r] = mk0 ? -1e30f : s0[r] * scale;
            p1[r] = mk1 ? -1e30f : s1[r] * scale;
            tmax[r] = fmaxf(p0[r], p1[r]);
        }
        #pragma unroll
        for (int off = 1; off < 16; off <<= 1)
            #pragma unroll
            for (int r = 0; r < 8; ++r)
                tmax[r] = fmaxf(tmax[r], __shfl_xor(tmax[r], off, 32));

        float corr[8], psum[8];
        #pragma unroll
        for (int r = 0; r < 8; ++r) {
            float mn = fmaxf(mrow[r], tmax[r]);
            corr[r]  = __expf(mrow[r] - mn);
            mrow[r]  = mn;
            p0[r]    = __expf(p0[r] - mn);   // masked lanes underflow to 0
            p1[r]    = __expf(p1[r] - mn);
            psum[r]  = p0[r] + p1[r];
        }
        #pragma unroll
        for (int off = 1; off < 16; off <<= 1)
            #pragma unroll
            for (int r = 0; r < 8; ++r)
                psum[r] += __shfl_xor(psum[r], off, 32);

        #pragma unroll
        for (int r = 0; r < 8; ++r) {
            lrow[r] = lrow[r] * corr[r] + psum[r];
            #pragma unroll
            for (int d = 0; d < 4; ++d) o[d][r] *= corr[r];
            P[(r + hi * 8) * 32 + m16]      = (half_t)p0[r];   // transpose via LDS
            P[(r + hi * 8) * 32 + 16 + m16] = (half_t)p1[r];
        }
        __syncthreads();

        // P as A-operand (16 queries x 32 keys)
        v16h pa;
        #pragma unroll
        for (int j = 0; j < 8; ++j) pa[j]     = P[m16 * 32 + hi * 8 + j];
        #pragma unroll
        for (int j = 0; j < 8; ++j) pa[8 + j] = P[m16 * 32 + 16 + hi * 8 + j];

        // V as B-operand per 16-wide d tile (all 32 key rows in range)
        #pragma unroll
        for (int d = 0; d < 4; ++d) {
            v16h bv;
            #pragma unroll
            for (int j = 0; j < 16; ++j)
                bv[j] = vb[(size_t)(kk + (j < 8 ? hi * 8 + j : 16 + hi * 8 + (j - 8))) * HD + d * 16 + m16];
            o[d] = __builtin_amdgcn_wmma_f32_16x16x32_f16(
                false, pa, false, bv, (short)0, o[d], false, false);
        }
        __syncthreads();
    }

    #pragma unroll
    for (int d = 0; d < 4; ++d)
        #pragma unroll
        for (int r = 0; r < 8; ++r) {
            int row = q0 + hi * 8 + r;
            float v = o[d][r] / lrow[r];
            O[((size_t)b * S + row) * HD + h * DD + d * 16 + m16] = (half_t)v;
        }
}

// ---------------------------------------------------------------------------
// LayerNorm (f32 in -> f16 out), one 256-thread block per 512-wide row
// ---------------------------------------------------------------------------
__global__ __launch_bounds__(256)
void layernorm_f16_kernel(const float* __restrict__ X,
                          const float* __restrict__ g,
                          const float* __restrict__ bt,
                          half_t* __restrict__ Y, int C)
{
    __shared__ float red[2][8];
    const int row = blockIdx.x;
    const int t   = threadIdx.x;
    const float* x = X + (size_t)row * C;
    float s = 0.f, s2 = 0.f;
    for (int c = t; c < C; c += 256) { float v = x[c]; s += v; s2 += v * v; }
    #pragma unroll
    for (int off = 16; off >= 1; off >>= 1) {
        s  += __shfl_xor(s,  off, 32);
        s2 += __shfl_xor(s2, off, 32);
    }
    if ((t & 31) == 0) { red[0][t >> 5] = s; red[1][t >> 5] = s2; }
    __syncthreads();
    float ts = 0.f, ts2 = 0.f;
    #pragma unroll
    for (int w = 0; w < 8; ++w) { ts += red[0][w]; ts2 += red[1][w]; }
    float mean = ts / C;
    float var  = ts2 / C - mean * mean;
    float inv  = rsqrtf(var + 1e-5f);
    half_t* y = Y + (size_t)row * C;
    for (int c = t; c < C; c += 256)
        y[c] = (half_t)((x[c] - mean) * inv * g[c] + bt[c]);
}

// ---------------------------------------------------------------------------
// elementwise helpers
// ---------------------------------------------------------------------------
__global__ void f32_to_f16_kernel(const float* __restrict__ src,
                                  half_t* __restrict__ dst, size_t n)
{
    size_t i = (size_t)blockIdx.x * blockDim.x + threadIdx.x;
    size_t st = (size_t)gridDim.x * blockDim.x;
    for (; i < n; i += st) dst[i] = (half_t)src[i];
}

__global__ void ctx_pad_f16_kernel(const float* __restrict__ ctx,
                                   half_t* __restrict__ dst, size_t total)
{
    size_t i = (size_t)blockIdx.x * blockDim.x + threadIdx.x;
    size_t st = (size_t)gridDim.x * blockDim.x;
    for (; i < total; i += st) {
        size_t b = i / ((size_t)TPAD * DIMC);
        size_t tk = (i / DIMC) % TPAD;
        size_t c = i % DIMC;
        dst[i] = (tk < TT) ? (half_t)ctx[((size_t)b * TT + tk) * DIMC + c]
                           : (half_t)0.f;
    }
}

__global__ void geglu_act_kernel(const half_t* __restrict__ G,
                                 half_t* __restrict__ FF, size_t total)
{
    size_t i = (size_t)blockIdx.x * blockDim.x + threadIdx.x;
    size_t st = (size_t)gridDim.x * blockDim.x;
    for (; i < total; i += st) {
        size_t row = i / 2048;
        size_t col = i % 2048;
        float y  = (float)G[row * 4096 + col];
        float gt = (float)G[row * 4096 + 2048 + col];
        float th = tanhf(gt * 0.7978845608f * (1.f + 0.044715f * gt * gt));
        FF[i] = (half_t)(y * 0.5f * gt * (1.f + th));
    }
}

// ---------------------------------------------------------------------------
// launcher
// ---------------------------------------------------------------------------
extern "C" void kernel_launch(void* const* d_in, const int* in_sizes, int n_in,
                              void* d_out, int out_size, void* d_ws, size_t ws_size,
                              hipStream_t stream)
{
    const float* x      = (const float*)d_in[0];
    const float* ctx    = (const float*)d_in[1];
    const float* ln1_g  = (const float*)d_in[2];
    const float* ln1_b  = (const float*)d_in[3];
    const float* wq1    = (const float*)d_in[4];
    const float* wk1    = (const float*)d_in[5];
    const float* wv1    = (const float*)d_in[6];
    const float* wo1    = (const float*)d_in[7];
    const float* bo1    = (const float*)d_in[8];
    const float* ln2_g  = (const float*)d_in[9];
    const float* ln2_b  = (const float*)d_in[10];
    const float* wq2    = (const float*)d_in[11];
    const float* wk2    = (const float*)d_in[12];
    const float* wv2    = (const float*)d_in[13];
    const float* wo2    = (const float*)d_in[14];
    const float* bo2    = (const float*)d_in[15];
    const float* ln3_g  = (const float*)d_in[16];
    const float* ln3_b  = (const float*)d_in[17];
    const float* gegw   = (const float*)d_in[18];
    const float* gegb   = (const float*)d_in[19];
    const float* outw   = (const float*)d_in[20];
    const float* outb   = (const float*)d_in[21];
    float* hbuf = (float*)d_out;                 // running residual stream

    const size_t ROWS = (size_t)BB * SS;         // 8192
    char* p = (char*)d_ws;
    auto take = [&](size_t bytes) -> half_t* {
        half_t* r = (half_t*)p;
        p += (bytes + 255) & ~(size_t)255;
        return r;
    };
    half_t* wq1h = take((size_t)DIMC * DIMC * 2);
    half_t* wk1h = take((size_t)DIMC * DIMC * 2);
    half_t* wv1h = take((size_t)DIMC * DIMC * 2);
    half_t* wo1h = take((size_t)DIMC * DIMC * 2);
    half_t* wq2h = take((size_t)DIMC * DIMC * 2);
    half_t* wk2h = take((size_t)DIMC * DIMC * 2);
    half_t* wv2h = take((size_t)DIMC * DIMC * 2);
    half_t* wo2h = take((size_t)DIMC * DIMC * 2);
    half_t* gegwh = take((size_t)DIMC * 4096 * 2);
    half_t* outwh = take((size_t)2048 * DIMC * 2);
    half_t* A16 = take(ROWS * DIMC * 2);
    half_t* Q16 = take(ROWS * DIMC * 2);
    half_t* K16 = take(ROWS * DIMC * 2);
    half_t* V16 = take(ROWS * DIMC * 2);
    half_t* O16 = take(ROWS * DIMC * 2);
    half_t* C16  = take((size_t)BB * TPAD * DIMC * 2);
    half_t* KC16 = take((size_t)BB * TPAD * DIMC * 2);
    half_t* VC16 = take((size_t)BB * TPAD * DIMC * 2);
    half_t* G16  = take(ROWS * 4096 * 2);
    half_t* FF16 = Q16;   // reuse Q/K/V/O region (contiguous) after attention

    auto cvt = [&](const float* src, half_t* dst, size_t n) {
        unsigned g = (unsigned)((n + 255) / 256);
        if (g > 4096) g = 4096;
        f32_to_f16_kernel<<<g, 256, 0, stream>>>(src, dst, n);
    };
    auto gemmH = [&](const half_t* A, const half_t* W, int M, int N, int K, half_t* oH) {
        dim3 grid(N / TBN, M / TBM);
        gemm_f16_wmma_kernel<false, false, false><<<grid, 256, 0, stream>>>(
            A, W, M, N, K, nullptr, nullptr, nullptr, oH);
    };
    auto gemmBiasH = [&](const half_t* A, const half_t* W, int M, int N, int K,
                         const float* bias, half_t* oH) {
        dim3 grid(N / TBN, M / TBM);
        gemm_f16_wmma_kernel<true, false, false><<<grid, 256, 0, stream>>>(
            A, W, M, N, K, bias, nullptr, nullptr, oH);
    };
    auto gemmBiasResF = [&](const half_t* A, const half_t* W, int M, int N, int K,
                            const float* bias, const float* resid, float* oF) {
        dim3 grid(N / TBN, M / TBM);
        gemm_f16_wmma_kernel<true, true, true><<<grid, 256, 0, stream>>>(
            A, W, M, N, K, bias, resid, oF, nullptr);
    };

    cvt(wq1, wq1h, (size_t)DIMC * DIMC);
    cvt(wk1, wk1h, (size_t)DIMC * DIMC);
    cvt(wv1, wv1h, (size_t)DIMC * DIMC);
    cvt(wo1, wo1h, (size_t)DIMC * DIMC);
    cvt(wq2, wq2h, (size_t)DIMC * DIMC);
    cvt(wk2, wk2h, (size_t)DIMC * DIMC);
    cvt(wv2, wv2h, (size_t)DIMC * DIMC);
    cvt(wo2, wo2h, (size_t)DIMC * DIMC);
    cvt(gegw, gegwh, (size_t)DIMC * 4096);
    cvt(outw, outwh, (size_t)2048 * DIMC);

    // ---- self-attention ----
    layernorm_f16_kernel<<<(unsigned)ROWS, 256, 0, stream>>>(x, ln1_g, ln1_b, A16, DIMC);
    gemmH(A16, wq1h, (int)ROWS, DIMC, DIMC, Q16);
    gemmH(A16, wk1h, (int)ROWS, DIMC, DIMC, K16);
    gemmH(A16, wv1h, (int)ROWS, DIMC, DIMC, V16);
    {
        dim3 grid(SS / 16, HH, BB);
        flash_attn_wmma_kernel<<<grid, 32, 0, stream>>>(Q16, K16, V16, O16,
                                                        SS, SS, SS, 0.125f);
    }
    gemmBiasResF(O16, wo1h, (int)ROWS, DIMC, DIMC, bo1, x, hbuf);    // h1

    // ---- cross-attention ----
    layernorm_f16_kernel<<<(unsigned)ROWS, 256, 0, stream>>>(hbuf, ln2_g, ln2_b, A16, DIMC);
    gemmH(A16, wq2h, (int)ROWS, DIMC, DIMC, Q16);
    {
        size_t total = (size_t)BB * TPAD * DIMC;
        ctx_pad_f16_kernel<<<(unsigned)((total + 255) / 256), 256, 0, stream>>>(ctx, C16, total);
    }
    gemmH(C16, wk2h, BB * TPAD, DIMC, DIMC, KC16);
    gemmH(C16, wv2h, BB * TPAD, DIMC, DIMC, VC16);
    {
        dim3 grid(SS / 16, HH, BB);
        flash_attn_wmma_kernel<<<grid, 32, 0, stream>>>(Q16, KC16, VC16, O16,
                                                        SS, TPAD, TT, 0.125f);
    }
    gemmBiasResF(O16, wo2h, (int)ROWS, DIMC, DIMC, bo2, hbuf, hbuf); // h2

    // ---- GEGLU FFN ----
    layernorm_f16_kernel<<<(unsigned)ROWS, 256, 0, stream>>>(hbuf, ln3_g, ln3_b, A16, DIMC);
    gemmBiasH(A16, gegwh, (int)ROWS, 4096, DIMC, gegb, G16);
    geglu_act_kernel<<<4096, 256, 0, stream>>>(G16, FF16, ROWS * 2048);
    gemmBiasResF(FF16, outwh, (int)ROWS, DIMC, 2048, outb, hbuf, hbuf); // out
}
